// LowRankSparseHyperedgeGen_25099788878232
// MI455X (gfx1250) — matrix operand order repaired and checked by reading, not compiled, not run on hardware
//
#include <hip/hip_runtime.h>
#include <hip/hip_bf16.h>
#include <math.h>

// ---------------------------------------------------------------------------
// LowRankSparseHyperedgeGen for MI455X (gfx1250, wave32, WMMA)
// B=8, N=4096, D=256, E=2048, R=16, H=4, dh=64, kE=512
// logits = (X@Wp.T+bp) @ proto^T / 32  (head split collapses: h,d both summed)
// Dominant GEMMs run on v_wmma_f32_16x16x32_f16; rest is tiny fp32 work.
// ---------------------------------------------------------------------------

typedef __attribute__((ext_vector_type(16))) _Float16 v16h;
typedef __attribute__((ext_vector_type(8)))  _Float16 v8h;
typedef __attribute__((ext_vector_type(8)))  float    v8f;

#define B_  8
#define N_  4096
#define D_  256
#define E_  2048
#define R_  16
#define KE_ 512
#define CTXW 512           // 2*D
#define LOGSTRIDE 513      // 512 + 1 pad -> distinct LDS banks across rows

// ---- WMMA fragment loaders (CDNA5 16-bit layouts, CDNA5 ISA) ----
// A (16x32, MxK): lane m=lane&15; lanes 0-15 hold K= 0..7 (elems 0-7) and
// K=16..23 (elems 8-15); lanes 16-31 hold K=8..15 and K=24..31.
__device__ __forceinline__ v16h load_afrag(const _Float16* A, int lda,
                                           int mbase, int kbase, int lane) {
    const int row  = mbase + (lane & 15);
    const int koff = (lane >> 4) * 8;
    const v8h lo = *(const v8h*)(A + (size_t)row * lda + kbase + koff);
    const v8h hi = *(const v8h*)(A + (size_t)row * lda + kbase + 16 + koff);
    v16h r;
#pragma unroll
    for (int j = 0; j < 8; ++j) { r[j] = lo[j]; r[8 + j] = hi[j]; }
    return r;
}

// B (32x16, KxN) for C = A*B^T with Brow = row-major [n, k] matrix:
// lane holds column n=lane&15; lanes 0-15 hold K=0..7 (elems 0-7) and
// K=16..23 (elems 8-15); lanes 16-31 hold K=8..15 and K=24..31.
__device__ __forceinline__ v16h load_bfrag(const _Float16* Brow, int ldb,
                                           int nbase, int kbase, int lane) {
    const int col = nbase + (lane & 15);
    const int kb  = kbase + (lane >> 4) * 8;
    const v8h lo = *(const v8h*)(Brow + (size_t)col * ldb + kb);
    const v8h hi = *(const v8h*)(Brow + (size_t)col * ldb + kb + 16);
    v16h r;
#pragma unroll
    for (int j = 0; j < 8; ++j) { r[j] = lo[j]; r[8 + j] = hi[j]; }
    return r;
}

// ---------------------------------------------------------------------------
// 1) ctx = [mean over N | max over N]  -> [B, 2D]
__global__ __launch_bounds__(256) void ctx_kernel(const float* __restrict__ X,
                                                  float* __restrict__ ctx) {
    __shared__ float ssum[256];
    __shared__ float smax[256];
    const int col = blockIdx.x;          // b*256 + d
    const int b = col >> 8, d = col & 255;
    const float* p = X + (size_t)b * N_ * D_ + d;
    float s = 0.f, m = -INFINITY;
    for (int n = threadIdx.x; n < N_; n += 256) {
        float v = p[(size_t)n * D_];
        s += v; m = fmaxf(m, v);
    }
    ssum[threadIdx.x] = s; smax[threadIdx.x] = m;
    __syncthreads();
    for (int off = 128; off > 0; off >>= 1) {
        if (threadIdx.x < off) {
            ssum[threadIdx.x] += ssum[threadIdx.x + off];
            smax[threadIdx.x]  = fmaxf(smax[threadIdx.x], smax[threadIdx.x + off]);
        }
        __syncthreads();
    }
    if (threadIdx.x == 0) {
        ctx[b * CTXW + d]       = ssum[0] * (1.0f / N_);
        ctx[b * CTXW + D_ + d]  = smax[0];
    }
}

// 2) V_dyn[b,r,d] = V[r,d] + bc[r*D+d] + ctx[b,:] . Wc[r*D+d,:]
__global__ __launch_bounds__(256) void vdyn_kernel(const float* __restrict__ V,
                                                   const float* __restrict__ Wc,
                                                   const float* __restrict__ bc,
                                                   const float* __restrict__ ctx,
                                                   float* __restrict__ Vdyn) {
    const int idx = blockIdx.x * 256 + threadIdx.x;   // < B*R*D = 32768
    const int b = idx >> 12, rd = idx & 4095;
    const float* c = ctx + b * CTXW;
    const float* w = Wc + (size_t)rd * CTXW;
    float acc = bc[rd];
    for (int j = 0; j < CTXW; ++j) acc += c[j] * w[j];
    Vdyn[idx] = V[rd] + acc;
}

// 3) sc[b,d] = ctx[b,:] . Ws[d,:] + bs[d]
__global__ __launch_bounds__(256) void sc_kernel(const float* __restrict__ Ws,
                                                 const float* __restrict__ bs,
                                                 const float* __restrict__ ctx,
                                                 float* __restrict__ sc) {
    const int idx = blockIdx.x * 256 + threadIdx.x;   // < B*D = 2048
    const int b = idx >> 8, d = idx & 255;
    const float* c = ctx + b * CTXW;
    const float* w = Ws + (size_t)d * CTXW;
    float acc = bs[d];
    for (int j = 0; j < CTXW; ++j) acc += c[j] * w[j];
    sc[idx] = acc;
}

// 4) z[b,r] = V_dyn[b,r,:] . sc[b,:]
__global__ __launch_bounds__(128) void z_kernel(const float* __restrict__ Vdyn,
                                                const float* __restrict__ sc,
                                                float* __restrict__ z) {
    const int t = threadIdx.x;            // < B*R = 128
    const int b = t >> 4, r = t & 15;
    const float* v = Vdyn + ((size_t)b * R_ + r) * D_;
    const float* s = sc + b * D_;
    float acc = 0.f;
    for (int d = 0; d < D_; ++d) acc += v[d] * s[d];
    z[t] = acc;
}

// 5) scores[b,e] = U[e,:].z[b,:] + proto_bias[e,:].sc[b,:]
__global__ __launch_bounds__(256) void scores_kernel(const float* __restrict__ U,
                                                     const float* __restrict__ pb,
                                                     const float* __restrict__ z,
                                                     const float* __restrict__ sc,
                                                     float* __restrict__ scores) {
    const int idx = blockIdx.x * 256 + threadIdx.x;   // < B*E = 16384
    const int b = idx >> 11, e = idx & 2047;
    float acc = 0.f;
    const float* u = U + (size_t)e * R_;
    const float* zz = z + b * R_;
#pragma unroll
    for (int r = 0; r < R_; ++r) acc += u[r] * zz[r];
    const float* pbe = pb + (size_t)e * D_;
    const float* s = sc + b * D_;
    for (int d = 0; d < D_; ++d) acc += pbe[d] * s[d];
    scores[idx] = acc;
}

// 6) top-k via exact rank counting (matches lax.top_k: descending, ties->lower idx)
__global__ __launch_bounds__(256) void topk_kernel(const float* __restrict__ scores,
                                                   int* __restrict__ topk) {
    __shared__ float s[E_];
    const int b = blockIdx.x;
    for (int i = threadIdx.x; i < E_; i += 256) s[i] = scores[b * E_ + i];
    __syncthreads();
#pragma unroll
    for (int k = 0; k < E_ / 256; ++k) {
        const int e = threadIdx.x + k * 256;
        const float my = s[e];
        int rank = 0;
        for (int j = 0; j < E_; ++j) {
            float o = s[j];
            rank += (o > my) || (o == my && j < e);
        }
        if (rank < KE_) topk[b * KE_ + rank] = e;
    }
}

// 7) proto[b,k,d] = U[topk,:].V_dyn[b,:,d] + proto_bias[topk,d]  (stored f16)
__global__ __launch_bounds__(256) void proto_kernel(const float* __restrict__ U,
                                                    const float* __restrict__ pb,
                                                    const float* __restrict__ Vdyn,
                                                    const int* __restrict__ topk,
                                                    _Float16* __restrict__ protoF) {
    const int idx = blockIdx.x * 256 + threadIdx.x;   // < B*KE*D = 1M
    const int b = idx / (KE_ * D_);
    const int rem = idx - b * (KE_ * D_);
    const int k = rem >> 8, d = rem & 255;
    const int e = topk[b * KE_ + k];
    float acc = pb[(size_t)e * D_ + d];
    const float* u = U + (size_t)e * R_;
    const float* v = Vdyn + (size_t)b * R_ * D_ + d;
#pragma unroll
    for (int r = 0; r < R_; ++r) acc += u[r] * v[(size_t)r * D_];
    protoF[idx] = (_Float16)acc;
}

// f32 -> f16 cast
__global__ __launch_bounds__(256) void cast_f16_kernel(const float* __restrict__ in,
                                                       _Float16* __restrict__ out,
                                                       int n) {
    const int i = blockIdx.x * 256 + threadIdx.x;
    if (i < n) out[i] = (_Float16)in[i];
}

// 8) Xh = X @ Wp^T + bp  via WMMA.
//    Each wave owns one 16-col band (Nb) and four 16-row tiles: Wp fragments
//    are hoisted once (8 x v16h) and reused across the 4 M tiles (32 WMMAs).
__global__ __launch_bounds__(256) void xh_gemm_kernel(const _Float16* __restrict__ Xf,
                                                      const _Float16* __restrict__ Wpf,
                                                      const float* __restrict__ bp,
                                                      _Float16* __restrict__ Xh) {
    const int lane = threadIdx.x & 31;
    const int wid  = threadIdx.x >> 5;
    const int wave = blockIdx.x * 8 + wid;    // (B*N/64)*(D/16) = 8192 waves
    const int mq = wave >> 4;                 // 64-row quad index (512 of them)
    const int nt = wave & 15;                 // 16-col band
    const int Nb = nt * 16;

    v16h bfr[D_ / 32];
#pragma unroll
    for (int kk = 0; kk < D_ / 32; ++kk)
        bfr[kk] = load_bfrag(Wpf, D_, Nb, kk * 32, lane);

    const int n    = Nb + (lane & 15);
    const float bias = bp[n];

#pragma unroll
    for (int mi = 0; mi < 4; ++mi) {
        const int Mb = mq * 64 + mi * 16;
        v8f acc = {};
#pragma unroll
        for (int kk = 0; kk < D_ / 32; ++kk) {
            v16h a = load_afrag(Xf, D_, Mb, kk * 32, lane);
            acc = __builtin_amdgcn_wmma_f32_16x16x32_f16(false, a, false, bfr[kk],
                                                         (short)0, acc, false, false);
        }
        const int mrow = Mb + (lane >> 4) * 8;
#pragma unroll
        for (int r = 0; r < 8; ++r)
            Xh[(size_t)(mrow + r) * D_ + n] = (_Float16)(acc[r] + bias);
    }
}

// 9) fused logits (WMMA) + softmax + output writes.
//    One wave per 16-row strip of one batch. The 16x256 A strip is loaded
//    ONCE into registers (8 x v16h = 64 VGPRs) and reused across all 32
//    column tiles; per tile only the 8 proto B fragments are streamed.
__global__ __launch_bounds__(32) void logits_softmax_kernel(
        const _Float16* __restrict__ Xh,
        const _Float16* __restrict__ protoF,
        const int* __restrict__ topk,
        float* __restrict__ edge_idx_out,
        float* __restrict__ edge_w_out) {
    __shared__ float lds_logits[16 * LOGSTRIDE];
    __shared__ float lds_idx[KE_];
    const int lane = threadIdx.x;             // 0..31
    const int bid  = blockIdx.x;              // B * N/16 = 2048
    const int b  = bid >> 8;
    const int Mb = (bid & 255) * 16;
    const _Float16* A = Xh     + (size_t)b * N_  * D_;
    const _Float16* P = protoF + (size_t)b * KE_ * D_;

    for (int i = lane; i < KE_; i += 32)      // uniform trip count: EXEC stays full
        lds_idx[i] = (float)topk[b * KE_ + i];

    // hoist the invariant A strip (16 rows x K=256)
    v16h afr[D_ / 32];
#pragma unroll
    for (int kk = 0; kk < D_ / 32; ++kk)
        afr[kk] = load_afrag(A, D_, Mb, kk * 32, lane);

    const float inv_scale = 1.0f / 32.0f;     // 1/(sqrt(dh)*H)
#pragma unroll 1
    for (int ct = 0; ct < KE_ / 16; ++ct) {
        v8f acc = {};
#pragma unroll
        for (int kk = 0; kk < D_ / 32; ++kk) {
            v16h bf = load_bfrag(P, D_, ct * 16, kk * 32, lane);
            acc = __builtin_amdgcn_wmma_f32_16x16x32_f16(false, afr[kk], false, bf,
                                                         (short)0, acc, false, false);
        }
        const int n    = ct * 16 + (lane & 15);
        const int mrow = (lane >> 4) * 8;
#pragma unroll
        for (int r = 0; r < 8; ++r)
            lds_logits[(mrow + r) * LOGSTRIDE + n] = acc[r] * inv_scale;
    }
    __syncthreads();                          // single-wave WG: lowers to s_nop

    // softmax: 2 lanes per row, each owns 256 contiguous columns
    const int row  = lane & 15;
    const int half = lane >> 4;
    const float* lrow = &lds_logits[row * LOGSTRIDE + half * 256];
    float mx = -INFINITY;
    for (int i = 0; i < 256; ++i) mx = fmaxf(mx, lrow[i]);
    mx = fmaxf(mx, __shfl_xor(mx, 16));
    float sum = 0.f;
    for (int i = 0; i < 256; ++i) sum += __expf(lrow[i] - mx);
    sum += __shfl_xor(sum, 16);
    const float inv = 1.0f / sum;

    const size_t base = ((size_t)b * N_ + Mb + row) * KE_ + half * 256;
    float* w  = edge_w_out  + base;
    float* ix = edge_idx_out + base;
    for (int i = 0; i < 256; ++i) {
        w[i]  = __expf(lrow[i] - mx) * inv;
        ix[i] = lds_idx[half * 256 + i];
    }
}

__global__ void write_e_kernel(float* out, int pos) { out[pos] = (float)E_; }

// ---------------------------------------------------------------------------
extern "C" void kernel_launch(void* const* d_in, const int* in_sizes, int n_in,
                              void* d_out, int out_size, void* d_ws, size_t ws_size,
                              hipStream_t stream) {
    const float* X  = (const float*)d_in[0];   // [B,N,D]
    const float* U  = (const float*)d_in[1];   // [E,R]
    const float* V  = (const float*)d_in[2];   // [R,D]
    const float* pb = (const float*)d_in[3];   // [E,D]
    const float* Wc = (const float*)d_in[4];   // [R*D, 2D]
    const float* bc = (const float*)d_in[5];   // [R*D]
    const float* Ws = (const float*)d_in[6];   // [D, 2D]
    const float* bs = (const float*)d_in[7];   // [D]
    const float* Wp = (const float*)d_in[8];   // [D, D]
    const float* bp = (const float*)d_in[9];   // [D]

    // workspace carve-up (256B aligned)
    char* ws = (char*)d_ws;
    size_t off = 0;
    auto take = [&](size_t bytes) {
        char* p = ws + off;
        off = (off + bytes + 255) & ~(size_t)255;
        return p;
    };
    float*    ctx    = (float*)   take((size_t)B_ * CTXW * 4);
    float*    Vdyn   = (float*)   take((size_t)B_ * R_ * D_ * 4);
    float*    sc     = (float*)   take((size_t)B_ * D_ * 4);
    float*    z      = (float*)   take((size_t)B_ * R_ * 4);
    float*    scores = (float*)   take((size_t)B_ * E_ * 4);
    int*      topk   = (int*)     take((size_t)B_ * KE_ * 4);
    _Float16* Xf16   = (_Float16*)take((size_t)B_ * N_ * D_ * 2);
    _Float16* Wpf16  = (_Float16*)take((size_t)D_ * D_ * 2);
    _Float16* Xhf16  = (_Float16*)take((size_t)B_ * N_ * D_ * 2);
    _Float16* protoF = (_Float16*)take((size_t)B_ * KE_ * D_ * 2);
    (void)ws_size;

    float* edge_idx_out = (float*)d_out;                       // [B,N,KE]
    float* edge_w_out   = edge_idx_out + (size_t)B_ * N_ * KE_;// [B,N,KE]

    // small dense pipeline (exact fp32)
    ctx_kernel   <<<B_ * D_, 256, 0, stream>>>(X, ctx);
    vdyn_kernel  <<<(B_ * R_ * D_) / 256, 256, 0, stream>>>(V, Wc, bc, ctx, Vdyn);
    sc_kernel    <<<(B_ * D_) / 256, 256, 0, stream>>>(Ws, bs, ctx, sc);
    z_kernel     <<<1, B_ * R_, 0, stream>>>(Vdyn, sc, z);
    scores_kernel<<<(B_ * E_) / 256, 256, 0, stream>>>(U, pb, z, sc, scores);
    topk_kernel  <<<B_, 256, 0, stream>>>(scores, topk);
    proto_kernel <<<(B_ * KE_ * D_) / 256, 256, 0, stream>>>(U, pb, Vdyn, topk, protoF);

    // f16 casts for WMMA GEMMs
    cast_f16_kernel<<<(B_ * N_ * D_) / 256, 256, 0, stream>>>(X, Xf16, B_ * N_ * D_);
    cast_f16_kernel<<<(D_ * D_) / 256, 256, 0, stream>>>(Wp, Wpf16, D_ * D_);

    // Xh = X @ Wp^T + bp  (8192 waves, 8 waves/block, 4 M-tiles per wave)
    xh_gemm_kernel<<<(B_ * N_ / 64) * (D_ / 16) / 8, 256, 0, stream>>>(Xf16, Wpf16, bp, Xhf16);

    // fused logits GEMM + softmax + writes
    logits_softmax_kernel<<<B_ * (N_ / 16), 32, 0, stream>>>(Xhf16, protoF, topk,
                                                             edge_idx_out, edge_w_out);

    write_e_kernel<<<1, 1, 0, stream>>>((float*)d_out, out_size - 1);
}
